// Qtran_57861799412373
// MI455X (gfx1250) — matrix operand order repaired and checked by reading, not compiled, not run on hardware
//
#include <hip/hip_runtime.h>
#include <hip/hip_bf16.h>
#include <stddef.h>

typedef _Float16 half_t;
typedef __attribute__((ext_vector_type(16))) _Float16 v16h;
typedef __attribute__((ext_vector_type(8)))  _Float16 v8h;
typedef __attribute__((ext_vector_type(8)))  float    v8f;

#define HDIM 64
#define NACT 21
#define BATCH 262144
#define WPB 2                  // waves per block (wave32)
#define TPB (WPB * 32)

// ---- output offsets (floats), tuple flattened in return order ----
constexpr size_t OQ1  = 0;
constexpr size_t OQ2  = (size_t)BATCH * NACT;
constexpr size_t OJ   = 2 * (size_t)BATCH * NACT;
constexpr size_t OJM  = OJ + BATCH;
constexpr size_t OV   = OJ + 2 * (size_t)BATCH;
constexpr size_t OQJ  = OJ + 3 * (size_t)BATCH;
constexpr size_t OQJM = OQJ + 2 * (size_t)BATCH * NACT;

// ---- workspace offsets (half elements): weights transposed [out][Kpad] f16 ----
constexpr int OFF_ENC2 = 0;              // 64x64
constexpr int OFF_QW1  = 4096;           // 64x64
constexpr int OFF_QW2  = 8192;           // 32x64  (N 21->32 pad)
constexpr int OFF_VW1  = 10240;          // 64x64
constexpr int OFF_L2W1 = 14336;          // 64x96  (K 85->96 pad)
constexpr int OFF_L2W2 = 20480;          // 64x64
constexpr int OFF_L2W3 = 24576;          // 64x64
constexpr int OFF_L3   = 28672;          // 64x64
constexpr int OFF_L4   = 32768;          // 32x64  (N 21->32 pad)
constexpr int OFF_QP1  = 34816;          // 128x64
constexpr int OFF_QP2  = 43008;          // 64x128
constexpr int OFF_QP3  = 51200;          // 64x64

#define ACT_NONE 0
#define ACT_RELU 1
#define ACT_ELU  2

// ================= fragment loaders (CDNA5 WMMA f16 layouts) =================
// A = WEIGHT tile (16x32 f16): rows = output features, from transposed [out][K] global.
// lanes 0-15 row m hold K 0..7 & 16..23; lanes 16-31 row m hold K 8..15 & 24..31
__device__ inline v16h load_frag_w(const half_t* wt, int ldk, int mBaseTile, int kOff, int lane) {
    const int m  = mBaseTile + (lane & 15);
    const int kb = kOff + ((lane < 16) ? 0 : 8);
    const v8h c0 = *(const v8h*)(wt + m * ldk + kb);         // global_load_b128
    const v8h c1 = *(const v8h*)(wt + m * ldk + kb + 16);    // global_load_b128
    v16h a;
#pragma unroll
    for (int i = 0; i < 8; ++i) { a[i] = c0[i]; a[i + 8] = c1[i]; }
    return a;
}

// B = ACTIVATION tile (32x16 f16): columns = samples, from [sample][feat] LDS staging.
// lanes 0-15 col n hold K 0..7 & 16..23; lanes 16-31 col n hold K 8..15 & 24..31
__device__ inline v16h load_frag_act(const half_t* buf, int ldk, int kOff, int lane) {
    const int n  = lane & 15;
    const int ks = kOff + ((lane < 16) ? 0 : 8);
    const v8h c0 = *(const v8h*)(buf + n * ldk + ks);        // ds_load_b128
    const v8h c1 = *(const v8h*)(buf + n * ldk + ks + 16);   // ds_load_b128
    v16h b;
#pragma unroll
    for (int i = 0; i < 8; ++i) { b[i] = c0[i]; b[i + 8] = c1[i]; }
    return b;
}

// ===== generic WMMA layer: D = W^T x Act. 16 samples, K->NOUT, activation, f16 out =====
// C fragment: lane = sample column, 8 contiguous output features -> one ds_store_b128.
template <int K, int NOUT, int ACT>
__device__ inline void layer_gemm(const half_t* __restrict__ bufIn,   // [16][K]
                                  const half_t* __restrict__ wt,      // [NOUT][K] f16
                                  const float*  __restrict__ bias,    // [NOUT] f32
                                  half_t* __restrict__ bufOut, int ldOut, int lane) {
    constexpr int NK = K / 32;
    v16h bfr[NK];                                // hoisted activation fragments
#pragma unroll
    for (int kt = 0; kt < NK; ++kt) bfr[kt] = load_frag_act(bufIn, K, kt * 32, lane);
    const int n     = lane & 15;                 // sample
    const int mBase = (lane < 16) ? 0 : 8;       // feature sub-tile
#pragma unroll
    for (int mt = 0; mt < NOUT; mt += 16) {
        v8f acc;
#pragma unroll
        for (int r = 0; r < 8; ++r) acc[r] = bias[mt + mBase + r];   // 2x b128 load
#pragma unroll
        for (int kt = 0; kt < NK; ++kt) {
            v16h a = load_frag_w(wt, K, mt, kt * 32, lane);
            acc = __builtin_amdgcn_wmma_f32_16x16x32_f16(false, a, false, bfr[kt],
                                                         (short)0, acc, false, false);
        }
        v8h outv;
#pragma unroll
        for (int r = 0; r < 8; ++r) {
            float v = acc[r];
            if (ACT == ACT_RELU) v = __builtin_fmaxf(v, 0.f);        // single v_max_num
            else if (ACT == ACT_ELU) v = v > 0.f ? v : (__expf(v) - 1.f);
            outv[r] = (half_t)v;
        }
        *(v8h*)(bufOut + n * ldOut + mt + mBase) = outv;             // ds_store_b128
    }
}

// head: K=64 -> 32 features (21 valid), no activation, f32 out into [16][32] LDS scratch
__device__ inline void layer_qhead(const half_t* __restrict__ bufIn,
                                   const half_t* __restrict__ wt,
                                   const float*  __restrict__ bias,
                                   float* __restrict__ fscr, int lane) {
    v16h b0 = load_frag_act(bufIn, 64, 0, lane);
    v16h b1 = load_frag_act(bufIn, 64, 32, lane);
    const int n     = lane & 15;
    const int mBase = (lane < 16) ? 0 : 8;
#pragma unroll
    for (int mt = 0; mt < 32; mt += 16) {
        v8f acc;
#pragma unroll
        for (int r = 0; r < 8; ++r)
            acc[r] = (mt + mBase + r < NACT) ? bias[mt + mBase + r] : 0.f;
        acc = __builtin_amdgcn_wmma_f32_16x16x32_f16(false, load_frag_w(wt, 64, mt, 0, lane),
                                                     false, b0, (short)0, acc, false, false);
        acc = __builtin_amdgcn_wmma_f32_16x16x32_f16(false, load_frag_w(wt, 64, mt, 32, lane),
                                                     false, b1, (short)0, acc, false, false);
#pragma unroll
        for (int r = 0; r < 8; ++r) fscr[n * 32 + mt + mBase + r] = acc[r];
    }
}

// enc layer1: K=2 -> 64 with ReLU, pure VALU (WMMA would waste 15/16 of the K tile)
__device__ inline void enc_layer1(const float* x, int sample0, int agent,
                                  const float* w1, const float* b1,
                                  half_t* bufOut, int lane) {
    const int m  = lane & 15;
    const int cb = (lane < 16) ? 0 : 32;
    const float x0 = x[(size_t)(sample0 + m) * 4 + agent * 2 + 0];
    const float x1 = x[(size_t)(sample0 + m) * 4 + agent * 2 + 1];
#pragma unroll
    for (int j = 0; j < 32; ++j) {
        const int nn = cb + j;
        float h = fmaf(x0, w1[nn], fmaf(x1, w1[HDIM + nn], b1[nn]));
        bufOut[m * HDIM + nn] = (half_t)__builtin_fmaxf(h, 0.f);
    }
}

// 64 -> 1 dot head: 2 lanes per sample, shfl_xor(16) cross-half reduce
__device__ inline float dot_head(const half_t* buf, const float* w, float bias, int lane) {
    const int m  = lane & 15;
    const int kb = (lane < 16) ? 0 : 32;
    float s = 0.f;
#pragma unroll
    for (int k = 0; k < 32; ++k)
        s = fmaf((float)buf[m * HDIM + kb + k], w[kb + k], s);
    s += __shfl_xor(s, 16, 32);
    return s + bias;   // valid in lanes 0..15
}

struct WaveSmem {
    __align__(16) half_t bufA[16 * 128];
    __align__(16) half_t bufB[16 * 128];
    __align__(16) half_t h1[16 * HDIM];
    __align__(16) half_t h2[16 * HDIM];
    __align__(16) half_t k1a[16 * HDIM];
    __align__(16) half_t k1b[16 * HDIM];
    float fscr[16 * 32];
    int   amax1[16];
    int   amax2[16];
    int   act1[16];
    int   act2[16];
};

struct Params {
    const float* x;
    const int*   b_act1; const int* b_act2;
    const float* enc_w1; const float* enc_b1; const float* enc_b2;
    const float* q_b1;   const float* q_b2;
    const float* v_b1;   const float* v_w2;   const float* v_b2;
    const float* l2_b1;  const float* l2_b2;  const float* l2_b3;
    const float* l3_b;   const float* l4_b;
    const float* qp_b1;  const float* qp_b2;  const float* qp_b3;
    const float* qp_w4;  const float* qp_b4;
    const half_t* ws;
    float* out;
};

__device__ inline void qhead_finish(const float* fscr, float* outQ, int sample0,
                                    int* amax, int lane) {
    for (int t = lane; t < 16 * NACT; t += 32) {
        const int s = t / NACT, c = t % NACT;
        outQ[(size_t)(sample0 + s) * NACT + c] = fscr[s * 32 + c];
    }
    if (lane < 16) {   // first-max tie rule == jnp.argmax
        float best = fscr[lane * 32]; int bi = 0;
#pragma unroll
        for (int c = 1; c < NACT; ++c) {
            const float v = fscr[lane * 32 + c];
            if (v > best) { best = v; bi = c; }
        }
        amax[lane] = bi;
    }
}

__device__ inline void build_e(const half_t* hbuf, const int* acts,
                               half_t* bufOut, int lane) {
    for (int t = lane; t < 16 * 96; t += 32) {
        const int m = t / 96, c = t % 96;
        float v;
        if (c < HDIM)             v = (float)hbuf[m * HDIM + c];
        else if (c < HDIM + NACT) v = (c - HDIM == acts[m]) ? 1.f : 0.f;
        else                      v = 0.f;
        bufOut[t] = (half_t)v;
    }
}

__global__ __launch_bounds__(TPB) void qtran_fused(Params p) {
    __shared__ WaveSmem smem[WPB];
    const int wave = threadIdx.x >> 5;
    const int lane = threadIdx.x & 31;
    WaveSmem& s = smem[wave];
    const int sample0 = (blockIdx.x * WPB + wave) * 16;

    const half_t* W_ENC2 = p.ws + OFF_ENC2;
    const half_t* W_Q1   = p.ws + OFF_QW1;
    const half_t* W_Q2   = p.ws + OFF_QW2;
    const half_t* W_V1   = p.ws + OFF_VW1;
    const half_t* W_L21  = p.ws + OFF_L2W1;
    const half_t* W_L22  = p.ws + OFF_L2W2;
    const half_t* W_L23  = p.ws + OFF_L2W3;
    const half_t* W_L3   = p.ws + OFF_L3;
    const half_t* W_L4   = p.ws + OFF_L4;
    const half_t* W_QP1  = p.ws + OFF_QP1;
    const half_t* W_QP2  = p.ws + OFF_QP2;
    const half_t* W_QP3  = p.ws + OFF_QP3;

    // ---- enc(x[:,0,:]) -> h1 ; enc(x[:,1,:]) -> h2 ----
    enc_layer1(p.x, sample0, 0, p.enc_w1, p.enc_b1, s.bufA, lane);
    layer_gemm<64, 64, ACT_RELU>(s.bufA, W_ENC2, p.enc_b2, s.h1, 64, lane);
    enc_layer1(p.x, sample0, 1, p.enc_w1, p.enc_b1, s.bufA, lane);
    layer_gemm<64, 64, ACT_RELU>(s.bufA, W_ENC2, p.enc_b2, s.h2, 64, lane);

    // ---- q1 = q_net(h1), q2 = q_net(h2) + argmax ----
    layer_gemm<64, 64, ACT_RELU>(s.h1, W_Q1, p.q_b1, s.bufB, 64, lane);
    layer_qhead(s.bufB, W_Q2, p.q_b2, s.fscr, lane);
    qhead_finish(s.fscr, p.out + OQ1, sample0, s.amax1, lane);
    layer_gemm<64, 64, ACT_RELU>(s.h2, W_Q1, p.q_b1, s.bufB, 64, lane);
    layer_qhead(s.bufB, W_Q2, p.q_b2, s.fscr, lane);
    qhead_finish(s.fscr, p.out + OQ2, sample0, s.amax2, lane);

    // ---- vtot = relu((h1+h2)@v_w1 + b)@v_w2 + b ----
    for (int t = lane; t < 16 * HDIM; t += 32)
        s.bufA[t] = (half_t)((float)s.h1[t] + (float)s.h2[t]);
    layer_gemm<64, 64, ACT_RELU>(s.bufA, W_V1, p.v_b1, s.bufB, 64, lane);
    {
        const float v = dot_head(s.bufB, p.v_w2, p.v_b2[0], lane);
        if (lane < 16) p.out[OV + sample0 + lane] = v;
    }

    // ---- actions for joint pass 0 ----
    if (lane < 16) {
        s.act1[lane] = p.b_act1[sample0 + lane];
        s.act2[lane] = p.b_act2[sample0 + lane];
    }

    for (int joint = 0; joint < 2; ++joint) {
        const int* a1 = joint ? s.amax1 : s.act1;
        const int* a2 = joint ? s.amax2 : s.act2;

        // lin2 per agent -> key1
        for (int ag = 0; ag < 2; ++ag) {
            const half_t* hh = ag ? s.h2 : s.h1;
            const int*    aa = ag ? a2 : a1;
            half_t*       kk = ag ? s.k1b : s.k1a;
            build_e(hh, aa, s.bufA, lane);
            layer_gemm<96, 64, ACT_RELU>(s.bufA, W_L21, p.l2_b1, s.bufB, 64, lane);
            layer_gemm<64, 64, ACT_RELU>(s.bufB, W_L22, p.l2_b2, s.bufA, 64, lane);
            layer_gemm<64, 64, ACT_NONE>(s.bufA, W_L23, p.l2_b3, kk, 64, lane);
        }

        // q_prime(mean(key1)) -> scalar
        for (int t = lane; t < 16 * HDIM; t += 32)
            s.bufA[t] = (half_t)(0.5f * ((float)s.k1a[t] + (float)s.k1b[t]));
        layer_gemm<64, 128, ACT_ELU>(s.bufA, W_QP1, p.qp_b1, s.bufB, 128, lane);
        layer_gemm<128, 64, ACT_ELU>(s.bufB, W_QP2, p.qp_b2, s.bufA, 64, lane);
        layer_gemm<64, 64, ACT_ELU>(s.bufA, W_QP3, p.qp_b3, s.bufB, 64, lane);
        {
            const float v = dot_head(s.bufB, p.qp_w4, p.qp_b4[0], lane);
            if (lane < 16) p.out[(joint ? OJM : OJ) + sample0 + lane] = v;
        }

        // q_jt: z_agent = h_agent + key1_other/2  (== key2 + mean(k1) - k1/2)
        float* oq = p.out + (joint ? OQJM : OQJ);
        for (int ag = 0; ag < 2; ++ag) {
            const half_t* hh = ag ? s.h2 : s.h1;
            const half_t* ko = ag ? s.k1a : s.k1b;
            for (int t = lane; t < 16 * HDIM; t += 32)
                s.bufA[t] = (half_t)((float)hh[t] + 0.5f * (float)ko[t]);
            layer_gemm<64, 64, ACT_RELU>(s.bufA, W_L3, p.l3_b, s.bufB, 64, lane);
            layer_qhead(s.bufB, W_L4, p.l4_b, s.fscr, lane);
            for (int t = lane; t < 16 * NACT; t += 32) {
                const int sm = t / NACT, c = t % NACT;
                oq[((size_t)(sample0 + sm) * 2 + ag) * NACT + c] = s.fscr[sm * 32 + c];
            }
        }
    }
}

// ================= weight prep: f32 [K][N] -> f16 transposed [Npad][Kpad] =================
__device__ inline void xpose_f16(const float* src, int K, int N, int Kpad, int Npad,
                                 half_t* dst, int tid, int nthreads) {
    for (int i = tid; i < Npad * Kpad; i += nthreads) {
        const int n = i / Kpad, k = i % Kpad;
        const float v = (n < N && k < K) ? src[(size_t)k * N + n] : 0.f;
        dst[i] = (half_t)v;
    }
}

__global__ void prep_weights(const float* enc_w2, const float* q_w1, const float* q_w2,
                             const float* v_w1, const float* l2_w1, const float* l2_w2,
                             const float* l2_w3, const float* l3_w, const float* l4_w,
                             const float* qp_w1, const float* qp_w2, const float* qp_w3,
                             half_t* ws) {
    const int tid = blockIdx.x * blockDim.x + threadIdx.x;
    const int nt  = gridDim.x * blockDim.x;
    xpose_f16(enc_w2,  64,  64,  64,  64, ws + OFF_ENC2, tid, nt);
    xpose_f16(q_w1,    64,  64,  64,  64, ws + OFF_QW1,  tid, nt);
    xpose_f16(q_w2,    64,  21,  64,  32, ws + OFF_QW2,  tid, nt);
    xpose_f16(v_w1,    64,  64,  64,  64, ws + OFF_VW1,  tid, nt);
    xpose_f16(l2_w1,   85,  64,  96,  64, ws + OFF_L2W1, tid, nt);
    xpose_f16(l2_w2,   64,  64,  64,  64, ws + OFF_L2W2, tid, nt);
    xpose_f16(l2_w3,   64,  64,  64,  64, ws + OFF_L2W3, tid, nt);
    xpose_f16(l3_w,    64,  64,  64,  64, ws + OFF_L3,   tid, nt);
    xpose_f16(l4_w,    64,  21,  64,  32, ws + OFF_L4,   tid, nt);
    xpose_f16(qp_w1,   64, 128,  64, 128, ws + OFF_QP1,  tid, nt);
    xpose_f16(qp_w2,  128,  64, 128,  64, ws + OFF_QP2,  tid, nt);
    xpose_f16(qp_w3,   64,  64,  64,  64, ws + OFF_QP3,  tid, nt);
}

extern "C" void kernel_launch(void* const* d_in, const int* in_sizes, int n_in,
                              void* d_out, int out_size, void* d_ws, size_t ws_size,
                              hipStream_t stream) {
    half_t* ws = (half_t*)d_ws;

    prep_weights<<<64, 256, 0, stream>>>(
        (const float*)d_in[5],   // enc_w2
        (const float*)d_in[7],   // q_w1
        (const float*)d_in[9],   // q_w2
        (const float*)d_in[11],  // v_w1
        (const float*)d_in[15],  // l2_w1
        (const float*)d_in[17],  // l2_w2
        (const float*)d_in[19],  // l2_w3
        (const float*)d_in[21],  // l3_w
        (const float*)d_in[23],  // l4_w
        (const float*)d_in[25],  // qp_w1
        (const float*)d_in[27],  // qp_w2
        (const float*)d_in[29],  // qp_w3
        ws);

    Params p;
    p.x      = (const float*)d_in[0];
    p.b_act1 = (const int*)d_in[1];
    p.b_act2 = (const int*)d_in[2];
    p.enc_w1 = (const float*)d_in[3];
    p.enc_b1 = (const float*)d_in[4];
    p.enc_b2 = (const float*)d_in[6];
    p.q_b1   = (const float*)d_in[8];
    p.q_b2   = (const float*)d_in[10];
    p.v_b1   = (const float*)d_in[12];
    p.v_w2   = (const float*)d_in[13];
    p.v_b2   = (const float*)d_in[14];
    p.l2_b1  = (const float*)d_in[16];
    p.l2_b2  = (const float*)d_in[18];
    p.l2_b3  = (const float*)d_in[20];
    p.l3_b   = (const float*)d_in[22];
    p.l4_b   = (const float*)d_in[24];
    p.qp_b1  = (const float*)d_in[26];
    p.qp_b2  = (const float*)d_in[28];
    p.qp_b3  = (const float*)d_in[30];
    p.qp_w4  = (const float*)d_in[31];
    p.qp_b4  = (const float*)d_in[32];
    p.ws     = ws;
    p.out    = (float*)d_out;

    const int blocks = BATCH / (16 * WPB);   // 8192
    qtran_fused<<<blocks, TPB, 0, stream>>>(p);
}
